// TransformerBlock_47665547051728
// MI455X (gfx1250) — matrix operand order repaired and checked by reading, not compiled, not run on hardware
//
#include <hip/hip_runtime.h>

#define BB   4
#define SS   2048
#define DD   1024
#define HH   16
#define DHD  64
#define TD   3072
#define FFD  2048
#define MROWS 8192

// Flip to 0 if the gfx1250 assembler rejects the TDM / async-LDS mnemonics.
#define USE_CDNA5_DMA 1

typedef __attribute__((ext_vector_type(16))) __bf16 v16bf;
typedef __attribute__((ext_vector_type(8)))  float  v8f;
typedef __attribute__((ext_vector_type(4)))  unsigned int u32x4;
typedef __attribute__((ext_vector_type(8)))  unsigned int u32x8;

union Frag { unsigned int u[8]; v16bf v; };

#define WMMA_BF16(A,B,C) \
  __builtin_amdgcn_wmma_f32_16x16x32_bf16(false,(A),false,(B),(short)0,(C),false,false)

#if USE_CDNA5_DMA
#if __has_builtin(__builtin_amdgcn_s_wait_asynccnt)
#define WAIT_ASYNC() __builtin_amdgcn_s_wait_asynccnt(0)
#else
#define WAIT_ASYNC() asm volatile("s_wait_asynccnt 0x0" ::: "memory")
#endif
#if __has_builtin(__builtin_amdgcn_s_wait_tensorcnt)
#define WAIT_TENSOR() __builtin_amdgcn_s_wait_tensorcnt(0)
#else
#define WAIT_TENSOR() asm volatile("s_wait_tensorcnt 0x0" ::: "memory")
#endif
#endif

__device__ __forceinline__ unsigned short f2bf(float f) {
  unsigned int u = __float_as_uint(f);
  u += 0x7FFFu + ((u >> 16) & 1u);           // round-to-nearest-even
  return (unsigned short)(u >> 16);
}
__device__ __forceinline__ unsigned int pack_bf2(float lo, float hi) {
  return (unsigned int)f2bf(lo) | ((unsigned int)f2bf(hi) << 16);
}

// ---- pure-VALU cross-lane reductions (v_permlane16 instead of ds_bpermute) ----
#if __has_builtin(__builtin_amdgcn_permlane16)
#define P16(x, lo, hi) __int_as_float(__builtin_amdgcn_permlane16( \
    __float_as_int(x), __float_as_int(x), (int)(lo), (int)(hi), false, false))
__device__ __forceinline__ float rmax16(float x) {
  x = fmaxf(x, P16(x, 0x67452301u, 0xEFCDAB89u));   // xor 1
  x = fmaxf(x, P16(x, 0x54761032u, 0xDCFE98BAu));   // xor 2
  x = fmaxf(x, P16(x, 0x32107654u, 0xBA98FEDCu));   // xor 4
  x = fmaxf(x, P16(x, 0xFEDCBA98u, 0x76543210u));   // xor 8
  return x;
}
__device__ __forceinline__ float rsum16(float x) {
  x += P16(x, 0x67452301u, 0xEFCDAB89u);
  x += P16(x, 0x54761032u, 0xDCFE98BAu);
  x += P16(x, 0x32107654u, 0xBA98FEDCu);
  x += P16(x, 0xFEDCBA98u, 0x76543210u);
  return x;
}
#else
__device__ __forceinline__ float rmax16(float x) {
  for (int off = 1; off < 16; off <<= 1) x = fmaxf(x, __shfl_xor(x, off, 16));
  return x;
}
__device__ __forceinline__ float rsum16(float x) {
  for (int off = 1; off < 16; off <<= 1) x += __shfl_xor(x, off, 16);
  return x;
}
#endif

#if __has_builtin(__builtin_amdgcn_permlane16) && __has_builtin(__builtin_amdgcn_permlanex16)
__device__ __forceinline__ float rsum32(float x) {
  x = rsum16(x);
  x += __int_as_float(__builtin_amdgcn_permlanex16(       // xor 16 (cross half)
      __float_as_int(x), __float_as_int(x), 0x76543210, 0xFEDCBA98, false, false));
  return x;
}
#else
__device__ __forceinline__ float rsum32(float x) {
  for (int off = 1; off < 32; off <<= 1) x += __shfl_xor(x, off, 32);
  return x;
}
#endif

// ---------------------------------------------------------------------------
// Prologue: f32 -> bf16 convert (paired stores)
// ---------------------------------------------------------------------------
__global__ __launch_bounds__(256)
void cvt_bf16_kernel(const float* __restrict__ in, unsigned short* __restrict__ out)
{
  int i = blockIdx.x * 256 + threadIdx.x;
  float2 t = *(const float2*)&in[2 * (size_t)i];
  *(unsigned int*)&out[2 * (size_t)i] = pack_bf2(t.x, t.y);
}

// Prologue: f32 [R,C] -> bf16 [C,R] tiled transpose (block 32x8)
__global__ __launch_bounds__(256)
void transpose_bf16_kernel(const float* __restrict__ in, unsigned short* __restrict__ out,
                           int R, int C)
{
  __shared__ float tile[32][33];
  int bx = blockIdx.x * 32, by = blockIdx.y * 32;
  int tx = threadIdx.x & 31, ty = threadIdx.x >> 5;
#pragma unroll
  for (int i = 0; i < 32; i += 8)
    tile[ty + i][tx] = in[(size_t)(by + ty + i) * C + bx + tx];
  __syncthreads();
#pragma unroll
  for (int i = 0; i < 32; i += 8)
    out[(size_t)(bx + ty + i) * R + by + tx] = f2bf(tile[tx][ty + i]);
}

// ---------------------------------------------------------------------------
// bf16 WMMA GEMM: C[M,N] = epi(A[M,K] @ Wt[N,K]^T + bias [, resid])
// Double-buffered LDS: async DMA for tile i+1 overlaps WMMA on tile i; the
// loop-head barrier provides both WAR safety and cross-wave visibility.
// Block 256 = 8 waves; 256x64 C tile; wave computes 32x64 (8 WMMA / k-step).
// EPI: 0 = bias -> f32, 1 = bias+silu -> bf16, 2 = bias+residual -> f32
// ---------------------------------------------------------------------------
template<int EPI>
__global__ __launch_bounds__(256)
void gemm_bf16(const unsigned short* __restrict__ A, const unsigned short* __restrict__ Wt,
               const float* __restrict__ bias, const float* __restrict__ resid,
               void* __restrict__ Cout, int M, int N, int K)
{
  __shared__ unsigned short As[2][256 * 40];   // [m][k], 64B rows padded to 80B
  __shared__ unsigned short Bs[2][64 * 40];    // [n][k], padded

  const int tid  = threadIdx.x;
  const int lane = tid & 31;
  const int w    = tid >> 5;
  const int g    = lane >> 4;
  const int nn   = lane & 15;
  const int m0   = blockIdx.y * 256;
  const int n0   = blockIdx.x * 64;

  v8f acc[2][4] = {};
#if USE_CDNA5_DMA
  const unsigned ldsA[2] = { (unsigned)(size_t)&As[0][0], (unsigned)(size_t)&As[1][0] };
  const unsigned ldsB[2] = { (unsigned)(size_t)&Bs[0][0], (unsigned)(size_t)&Bs[1][0] };
#endif

  auto stage = [&](int bi, int k0) {
#if USE_CDNA5_DMA
    int ar = tid >> 2, ch = tid & 3;   // A: 1024 x16B chunks (4/thr); B: 256 (1/thr)
#pragma unroll
    for (int i = 0; i < 4; ++i) {
      unsigned la = ldsA[bi] + (unsigned)((ar + 64 * i) * 80 + ch * 16);
      const unsigned short* pa = A + (size_t)(m0 + ar + 64 * i) * K + k0 + ch * 8;
      asm volatile("global_load_async_to_lds_b128 %0, %1, off" :: "v"(la), "v"(pa) : "memory");
    }
    unsigned lb = ldsB[bi] + (unsigned)(ar * 80 + ch * 16);
    const unsigned short* pb = Wt + (size_t)(n0 + ar) * K + k0 + ch * 8;
    asm volatile("global_load_async_to_lds_b128 %0, %1, off" :: "v"(lb), "v"(pb) : "memory");
#else
#pragma unroll
    for (int i = 0; i < 16; ++i) {          // A: 4096 dwords
      int e = tid + i * 256, r = e >> 4, cp = e & 15;
      *(unsigned int*)&As[bi][r * 40 + 2 * cp] =
          *(const unsigned int*)&A[(size_t)(m0 + r) * K + k0 + 2 * cp];
    }
#pragma unroll
    for (int i = 0; i < 4; ++i) {           // B: 1024 dwords
      int e = tid + i * 256, r = e >> 4, cp = e & 15;
      *(unsigned int*)&Bs[bi][r * 40 + 2 * cp] =
          *(const unsigned int*)&Wt[(size_t)(n0 + r) * K + k0 + 2 * cp];
    }
#endif
  };

  stage(0, 0);
  for (int k0 = 0; k0 < K; k0 += 32) {
    const int cur = (k0 >> 5) & 1;
#if USE_CDNA5_DMA
    WAIT_ASYNC();
#endif
    __syncthreads();                         // cur ready everywhere; nxt free to overwrite
    if (k0 + 32 < K) stage(cur ^ 1, k0 + 32);

    Frag fa0, fa1;
#pragma unroll
    for (int v = 0; v < 8; ++v) {
      int kb = 2 * v + 16 * g;   // A-layout K pattern
      fa0.u[v] = *(const unsigned int*)&As[cur][(32 * w + nn) * 40 + kb];
      fa1.u[v] = *(const unsigned int*)&As[cur][(32 * w + 16 + nn) * 40 + kb];
    }
#pragma unroll
    for (int tc = 0; tc < 4; ++tc) {
      Frag fb;
#pragma unroll
      for (int v = 0; v < 8; ++v)                      // B-layout K pattern
        fb.u[v] = *(const unsigned int*)&Bs[cur][(16 * tc + nn) * 40 + 2 * v + 16 * g];
      acc[0][tc] = WMMA_BF16(fa0.v, fb.v, acc[0][tc]);
      acc[1][tc] = WMMA_BF16(fa1.v, fb.v, acc[1][tc]);
    }
  }

  float* Cf = (float*)Cout;
  unsigned short* Cb = (unsigned short*)Cout;
#pragma unroll
  for (int rt = 0; rt < 2; ++rt) {
#pragma unroll
    for (int tc = 0; tc < 4; ++tc) {
#pragma unroll
      for (int r = 0; r < 8; ++r) {
        int row = m0 + 32 * w + 16 * rt + r + 8 * g;   // D layout: row = reg + 8*group
        int col = n0 + 16 * tc + nn;
        float v = acc[rt][tc][r] + bias[col];
        if (EPI == 1) {
          v = v / (1.0f + __expf(-v));                 // silu -> bf16 output
          Cb[(size_t)row * N + col] = f2bf(v);
        } else {
          if (EPI == 2) v += resid[(size_t)row * N + col];
          Cf[(size_t)row * N + col] = v;
        }
      }
    }
  }
}

// ---------------------------------------------------------------------------
// L2-normalize q/k per head, scale, xpos rotary; emit bf16 Q,K,V [B,H,S,dh].
// Q pre-scaled by 1/sqrt(dh).  Fast-math transcendentals.
// ---------------------------------------------------------------------------
__global__ __launch_bounds__(256)
void rope_norm_kernel(const float* __restrict__ qkv,
                      const float* __restrict__ q_scale,
                      const float* __restrict__ k_scale,
                      unsigned short* __restrict__ Qr,
                      unsigned short* __restrict__ Kr,
                      unsigned short* __restrict__ Vb)
{
  int gw   = blockIdx.x * 8 + (threadIdx.x >> 5);
  int lane = threadIdx.x & 31;
  int b    = gw >> 15;
  int rem  = gw & 32767;
  int s    = rem >> 4;
  int h    = rem & 15;

  const float* base = qkv + ((size_t)(b * SS + s)) * TD + h * 192;
  float q0 = base[lane],       q1 = base[lane + 32];
  float k0 = base[64 + lane],  k1 = base[96 + lane];
  float v0 = base[128 + lane], v1 = base[160 + lane];

  float nq = rsum32(q0 * q0 + q1 * q1);
  float nk = rsum32(k0 * k0 + k1 * k1);
  float rq = 1.0f / fmaxf(sqrtf(nq), 1e-12f);
  float rk = 1.0f / fmaxf(sqrtf(nk), 1e-12f);
  q0 *= rq * q_scale[lane]; q1 *= rq * q_scale[lane + 32];
  k0 *= rk * k_scale[lane]; k1 *= rk * k_scale[lane + 32];

  float j  = (float)lane;
  float fr = (float)s * __expf(-(j / 32.0f) * 9.210340371976184f); // 10000^(-j/32)
  float c, sn;
  __sincosf(fr, &sn, &c);
  float bse = (2.0f * j + 25.6f) / 89.6f;
  float pw  = ((float)s - (float)(SS / 2)) / 512.0f;
  float sc  = __expf(pw * __logf(bse));     // bse > 0 always
  float isc = 1.0f / sc;
  float qs  = sc * 0.125f;                  // fold 1/sqrt(64) into Q

  size_t o = ((size_t)(b * HH + h) * SS + s) * DHD;
  Qr[o + lane]      = f2bf((q0 * c - q1 * sn) * qs);
  Qr[o + lane + 32] = f2bf((q1 * c + q0 * sn) * qs);
  Kr[o + lane]      = f2bf((k0 * c - k1 * sn) * isc);
  Kr[o + lane + 32] = f2bf((k1 * c + k0 * sn) * isc);
  Vb[o + lane]      = f2bf(v0);
  Vb[o + lane + 32] = f2bf(v1);
}

// ---------------------------------------------------------------------------
// Flash attention: 8 waves/block, wave owns 16 query rows -> 128 q-rows share
// each K/V pass.  64-key blocks, double-buffered:
//   K tile : TDM tensor_load_to_lds (HW pads 128B rows to 144B)
//   V tile : per-lane global_load_async_to_lds_b128
// DMA for block i+1 overlaps the 16 WMMA + softmax of block i.  The P tile is
// wave-private, so no barrier is needed between softmax stores and P@V loads
// (per-wave DS ops are in-order).
// ---------------------------------------------------------------------------
__global__ __launch_bounds__(256)
void attn_kernel(const unsigned short* __restrict__ Qr,
                 const unsigned short* __restrict__ Kr,
                 const unsigned short* __restrict__ Vb,
                 float* __restrict__ out)
{
  __shared__ unsigned short Ks[2][64 * 72];   // [key][d], 128B rows padded to 144B
  __shared__ unsigned short Vs[2][64 * 72];   // [key][d]
  __shared__ unsigned short Ps[8 * 16 * 66];  // per-wave P tile [16][64]

  const int tid  = threadIdx.x;
  const int lane = tid & 31;
  const int w    = tid >> 5;
  const int g    = lane >> 4;
  const int nn   = lane & 15;
  const int bh   = blockIdx.y;
  const int b    = bh >> 4, h = bh & 15;
  const int qb   = blockIdx.x * 128 + w * 16;

  const unsigned short* qrow = Qr + ((size_t)bh * SS + qb + nn) * DHD;
  Frag qa0, qa1;
#pragma unroll
  for (int v = 0; v < 8; ++v) {
    int kb = 2 * v + 16 * g;
    qa0.u[v] = *(const unsigned int*)&qrow[kb];
    qa1.u[v] = *(const unsigned int*)&qrow[32 + kb];
  }

  v8f o[4] = {};
  float mrow[8], lrow[8];
#pragma unroll
  for (int r = 0; r < 8; ++r) { mrow[r] = -1e30f; lrow[r] = 0.f; }

  const unsigned short* kg = Kr + (size_t)bh * SS * DHD;
  const unsigned short* vg = Vb + (size_t)bh * SS * DHD;

#if USE_CDNA5_DMA
  const unsigned ldsK[2] = { (unsigned)(size_t)&Ks[0][0], (unsigned)(size_t)&Ks[1][0] };
  const unsigned ldsV[2] = { (unsigned)(size_t)&Vs[0][0], (unsigned)(size_t)&Vs[1][0] };
  // D# group1: data_size=2B, pad_enable, pad_interval=32dw(128B), pad_amount=4dw(16B)
  // dims: tensor 64 x 64 (= tile), row stride 64 elems
  const u32x8 g1 = { 0x07110000u, (64u << 16), (64u << 16), (64u << 16),
                     64u, 64u, 0u, 0u };
#endif

  auto stageKV = [&](int bi, int kb0) {
#if USE_CDNA5_DMA
    if (w == 0) {
      unsigned long long ga = (unsigned long long)(size_t)(kg + (size_t)kb0 * DHD);
      u32x4 g0 = { 1u, ldsK[bi], (unsigned)ga, (unsigned)(ga >> 32) | 0x80000000u };
      asm volatile("tensor_load_to_lds %0, %1" :: "s"(g0), "s"(g1) : "memory");
    }
    int vr = tid >> 3, ch = tid & 7;                   // 64 rows x 8 x 16B chunks
    unsigned l0 = ldsV[bi] + (unsigned)(vr * 144 + ch * 16);
    const unsigned short* p0 = vg + (size_t)(kb0 + vr) * DHD + ch * 8;
    asm volatile("global_load_async_to_lds_b128 %0, %1, off" :: "v"(l0), "v"(p0) : "memory");
    unsigned l1 = l0 + 32u * 144u;
    const unsigned short* p1 = p0 + 32 * DHD;
    asm volatile("global_load_async_to_lds_b128 %0, %1, off" :: "v"(l1), "v"(p1) : "memory");
#else
#pragma unroll
    for (int i = 0; i < 8; ++i) {
      int e = tid + i * 256, kr = e >> 5, cp = e & 31;
      *(unsigned int*)&Ks[bi][kr * 72 + 2 * cp] =
          *(const unsigned int*)&kg[(size_t)(kb0 + kr) * DHD + 2 * cp];
      *(unsigned int*)&Vs[bi][kr * 72 + 2 * cp] =
          *(const unsigned int*)&vg[(size_t)(kb0 + kr) * DHD + 2 * cp];
    }
#endif
  };

  stageKV(0, 0);
  for (int kb0 = 0; kb0 < SS; kb0 += 64) {
    const int cur = (kb0 >> 6) & 1;
#if USE_CDNA5_DMA
    WAIT_ASYNC();
    if (w == 0) WAIT_TENSOR();
#endif
    __syncthreads();                         // cur ready everywhere; nxt free to overwrite
    if (kb0 + 64 < SS) stageKV(cur ^ 1, kb0 + 64);

    // logits: S[16q x 64k] = Q @ K^T ; 4 key sub-tiles x 2 d-chunks
    v8f l[4];
#pragma unroll
    for (int jj = 0; jj < 4; ++jj) {
      Frag f0, f1;
#pragma unroll
      for (int v = 0; v < 8; ++v) {
        int doff = 16 * g + 2 * v;
        f0.u[v] = *(const unsigned int*)&Ks[cur][(16 * jj + nn) * 72 + doff];
        f1.u[v] = *(const unsigned int*)&Ks[cur][(16 * jj + nn) * 72 + 32 + doff];
      }
      v8f z = {};
      z = WMMA_BF16(qa0.v, f0.v, z);
      l[jj] = WMMA_BF16(qa1.v, f1.v, z);
    }

    // online softmax (scale pre-folded into Q); row = reg + 8*lane-group
#pragma unroll
    for (int r = 0; r < 8; ++r) {
      float x0 = l[0][r], x1 = l[1][r], x2 = l[2][r], x3 = l[3][r];
      float mx = rmax16(fmaxf(fmaxf(x0, x1), fmaxf(x2, x3)));
      float mn = fmaxf(mrow[r], mx);
      float al = __expf(mrow[r] - mn);
      mrow[r] = mn;
      float p0 = __expf(x0 - mn), p1 = __expf(x1 - mn);
      float p2 = __expf(x2 - mn), p3 = __expf(x3 - mn);
      float ps = rsum16((p0 + p1) + (p2 + p3));
      lrow[r] = lrow[r] * al + ps;
#pragma unroll
      for (int c = 0; c < 4; ++c) o[c][r] *= al;
      int pb = (w * 16 + r + 8 * g) * 66;
      Ps[pb + nn]      = f2bf(p0);
      Ps[pb + 16 + nn] = f2bf(p1);
      Ps[pb + 32 + nn] = f2bf(p2);
      Ps[pb + 48 + nn] = f2bf(p3);
    }
    // no barrier: Ps slice is wave-private, per-wave DS ops are in-order

    // O[16x64] += P[16x64] @ V[64x64]  (two K=32 chunks)
    Frag pa0, pa1;
#pragma unroll
    for (int v = 0; v < 8; ++v) {
      int kb = 2 * v + 16 * g;
      pa0.u[v] = *(const unsigned int*)&Ps[(w * 16 + nn) * 66 + kb];
      pa1.u[v] = *(const unsigned int*)&Ps[(w * 16 + nn) * 66 + 32 + kb];
    }
#pragma unroll
    for (int c = 0; c < 4; ++c) {
      Frag fv0, fv1;
#pragma unroll
      for (int v = 0; v < 8; ++v) {
        int kk = 2 * v + 16 * g;                       // B K-dim = key (vertical pair)
        fv0.u[v] = (unsigned)Vs[cur][kk * 72 + 16 * c + nn] |
                   ((unsigned)Vs[cur][(kk + 1) * 72 + 16 * c + nn] << 16);
        fv1.u[v] = (unsigned)Vs[cur][(kk + 32) * 72 + 16 * c + nn] |
                   ((unsigned)Vs[cur][(kk + 33) * 72 + 16 * c + nn] << 16);
      }
      o[c] = WMMA_BF16(pa0.v, fv0.v, o[c]);
      o[c] = WMMA_BF16(pa1.v, fv1.v, o[c]);
    }
  }

  // finalize: out layout [B,S,H,dh]
#pragma unroll
  for (int r = 0; r < 8; ++r) {
    int q = qb + r + 8 * g;
    float inv = 1.0f / lrow[r];
    float* op = out + ((size_t)(b * SS + q) * HH + h) * DHD;
#pragma unroll
    for (int c = 0; c < 4; ++c) op[16 * c + nn] = o[c][r] * inv;
  }
}

// ---------------------------------------------------------------------------
// LayerNorm over D=1024; emits f32 (for residual) and bf16 (FFN1 A-matrix)
// ---------------------------------------------------------------------------
__global__ __launch_bounds__(256)
void ln_kernel(const float* __restrict__ in, const float* __restrict__ gam,
               const float* __restrict__ bet, float* __restrict__ x,
               unsigned short* __restrict__ xb)
{
  __shared__ float red[16];
  int row = blockIdx.x, t = threadIdx.x;
  const float* p = in + (size_t)row * DD;
  float v[4]; float s = 0.f, ss = 0.f;
#pragma unroll
  for (int i = 0; i < 4; ++i) { v[i] = p[t + 256 * i]; s += v[i]; ss += v[i] * v[i]; }
  s = rsum32(s); ss = rsum32(ss);
  int w = t >> 5, lane = t & 31;
  if (lane == 0) { red[w] = s; red[8 + w] = ss; }
  __syncthreads();
  if (t == 0) {
    float S1 = 0, S2 = 0;
    for (int i = 0; i < 8; ++i) { S1 += red[i]; S2 += red[8 + i]; }
    float mu = S1 / (float)DD;
    red[0] = mu;
    red[1] = rsqrtf(S2 / (float)DD - mu * mu + 1e-5f);
  }
  __syncthreads();
  float mu = red[0], rstd = red[1];
#pragma unroll
  for (int i = 0; i < 4; ++i) {
    int idx = t + 256 * i;
    float o = (v[i] - mu) * rstd * gam[idx] + bet[idx];
    x[(size_t)row * DD + idx]  = o;
    xb[(size_t)row * DD + idx] = f2bf(o);
  }
}

// ---------------------------------------------------------------------------
extern "C" void kernel_launch(void* const* d_in, const int* in_sizes, int n_in,
                              void* d_out, int out_size, void* d_ws, size_t ws_size,
                              hipStream_t stream)
{
  (void)in_sizes; (void)n_in; (void)out_size; (void)ws_size;
  const float* Q       = (const float*)d_in[0];
  const float* Wqkv    = (const float*)d_in[3];
  const float* bqkv    = (const float*)d_in[4];
  const float* q_scale = (const float*)d_in[5];
  const float* k_scale = (const float*)d_in[6];
  const float* ln_g    = (const float*)d_in[7];
  const float* ln_b    = (const float*)d_in[8];
  const float* W1      = (const float*)d_in[9];
  const float* b1      = (const float*)d_in[10];
  const float* W2      = (const float*)d_in[11];
  const float* b2      = (const float*)d_in[12];
  float* outp = (float*)d_out;

  // Workspace (~182 MB), regions reused once producers are dead:
  //   [0, 100.7M)      qkv f32        (gemm0 -> rope)
  //   [0, 33.6M)       attnout f32    (attn -> ln, over dead qkv)
  //   [0, 33.6M)       ffh bf16       (ffn1 -> ffn2, over dead attnout)
  //   [33.6M, 50.4M)   x bf16         (ln -> ffn1, over dead qkv)
  //   [67.1M, 100.7M)  x f32          (ln -> ffn2, over dead qkv)
  //   [100.7M, 151M)   Qr/Kr/Vb bf16  (rope -> attn)
  //   [151M, 182.5M)   Qbf + transposed bf16 weights (prologue -> gemms)
  char* ws = (char*)d_ws;
  float*          qkvbuf  = (float*)ws;
  float*          attnout = (float*)ws;
  unsigned short* ffhb    = (unsigned short*)ws;
  unsigned short* xbf     = (unsigned short*)(ws + 33554432);
  float*          xbuf    = (float*)(ws + 67108864);
  unsigned short* Qr      = (unsigned short*)(ws + 100663296);
  unsigned short* Kr      = Qr + (size_t)BB * HH * SS * DHD;
  unsigned short* Vb      = Kr + (size_t)BB * HH * SS * DHD;
  unsigned short* Qbf     = (unsigned short*)(ws + 150994944);
  unsigned short* WqkvT   = (unsigned short*)(ws + 167772160);
  unsigned short* W1T     = (unsigned short*)(ws + 174063616);
  unsigned short* W2T     = (unsigned short*)(ws + 178257920);

  // Prologue: one-time bf16 conversion / weight transposition
  cvt_bf16_kernel<<<(MROWS * DD / 2) / 256, 256, 0, stream>>>(Q, Qbf);
  transpose_bf16_kernel<<<dim3(TD / 32, DD / 32), 256, 0, stream>>>(Wqkv, WqkvT, DD, TD);
  transpose_bf16_kernel<<<dim3(FFD / 32, DD / 32), 256, 0, stream>>>(W1, W1T, DD, FFD);
  transpose_bf16_kernel<<<dim3(DD / 32, FFD / 32), 256, 0, stream>>>(W2, W2T, FFD, DD);

  gemm_bf16<0><<<dim3(TD / 64, MROWS / 256), 256, 0, stream>>>(
      Qbf, WqkvT, bqkv, nullptr, qkvbuf, MROWS, TD, DD);
  rope_norm_kernel<<<BB * SS * HH / 8, 256, 0, stream>>>(
      qkvbuf, q_scale, k_scale, Qr, Kr, Vb);
  attn_kernel<<<dim3(SS / 128, BB * HH), 256, 0, stream>>>(Qr, Kr, Vb, attnout);
  ln_kernel<<<MROWS, 256, 0, stream>>>(attnout, ln_g, ln_b, xbuf, xbf);
  gemm_bf16<1><<<dim3(FFD / 64, MROWS / 256), 256, 0, stream>>>(
      xbf, W1T, b1, nullptr, ffhb, MROWS, FFD, DD);
  gemm_bf16<2><<<dim3(DD / 64, MROWS / 256), 256, 0, stream>>>(
      ffhb, W2T, b2, xbuf, outp, MROWS, DD, FFD);
}
